// PytorchMambaBlock_83365315215989
// MI455X (gfx1250) — matrix operand order repaired and checked by reading, not compiled, not run on hardware
//
#include <hip/hip_runtime.h>
#include <hip/hip_bf16.h>

// ---------------------------------------------------------------------------
// Mamba block forward for MI455X (gfx1250, wave32, WMMA + Tensor Data Mover).
// GEMMs: bf16 v_wmma_f32_16x16x32_bf16, 128x128 block tile, 32x64 wave tile,
//        double-buffered LDS filled by TDM (tensor_load_to_lds, TENSORcnt).
// Scan:  3-pass chunked parallel scan (chunk=128), exp via v_exp_f32.
// ---------------------------------------------------------------------------

typedef __attribute__((ext_vector_type(16))) __bf16 v16bf;
typedef __attribute__((ext_vector_type(8)))  float  v8f;
typedef __attribute__((ext_vector_type(4)))  unsigned int u32x4;
typedef __attribute__((ext_vector_type(8)))  int  i32x8;
typedef __attribute__((ext_vector_type(4)))  int  i32x4;

union FragAB { unsigned int u[8]; v16bf v; };
union FragC  { v8f v; float f[8]; };

__device__ __forceinline__ unsigned short f32_to_bf16(float x) {
  unsigned int u = __builtin_bit_cast(unsigned int, x);
  unsigned int r = (u + 0x7FFFu + ((u >> 16) & 1u)) >> 16;   // round-nearest-even
  return (unsigned short)r;
}

__device__ __forceinline__ float silu_f(float x) {
  return x / (1.0f + __expf(-x));
}

__device__ __forceinline__ unsigned lds_off32(const void* p) {
  // LDS aperture: hardware uses addr[31:0] as the LDS byte offset.
  return (unsigned)(unsigned long long)(size_t)p;
}

// Issue a 2-D bf16 tile load via the Tensor Data Mover.
// tile_d0 (row length, elems) x tile_d1 (rows); row stride in tensor = stride0.
__device__ __forceinline__ void tdm_load_tile_bf16(
    unsigned lds_addr, const unsigned short* gaddr,
    unsigned tensor_d0, unsigned tensor_d1,
    unsigned tile_d0, unsigned tile_d1, unsigned stride0) {
  unsigned long long ga = (unsigned long long)(size_t)gaddr;
  u32x4 g0;
  g0[0] = 1u;                                            // count=1, flags 0
  g0[1] = lds_addr;                                      // [63:32] lds_addr
  g0[2] = (unsigned)(ga & 0xFFFFFFFFull);                // [95:64] global lo
  g0[3] = ((unsigned)(ga >> 32) & 0x01FFFFFFu)           // [120:96] global hi
          | 0x80000000u;                                 // [127:126] type=2
  i32x8 g1;
  g1[0] = (int)0x00010000u;                              // data_size=2B, mask=0
  g1[1] = (int)(tensor_d0 << 16);                        // tensor_dim0[15:0]
  g1[2] = (int)((tensor_d0 >> 16) | (tensor_d1 << 16));  // dim0 hi | dim1 lo
  g1[3] = (int)((tensor_d1 >> 16) | (tile_d0 << 16));    // dim1 hi | tile_dim0
  g1[4] = (int)(tile_d1 & 0xFFFFu);                      // tile_dim1, tile_dim2=0
  g1[5] = (int)stride0;                                  // tensor_dim0_stride lo
  g1[6] = 0;                                             // stride0 hi | stride1 lo
  g1[7] = 0;                                             // stride1 hi
  i32x4 z4 = {0, 0, 0, 0};                               // groups 2/3 unused (2-D)
  i32x8 z8 = {0, 0, 0, 0, 0, 0, 0, 0};
  __builtin_amdgcn_tensor_load_to_lds(g0, g1, z4, z4, z8, 0);
}

// ----------------------------- cast f32 -> bf16 ----------------------------
__global__ __launch_bounds__(256) void cast_f32_bf16_kernel(
    const float* __restrict__ in, unsigned short* __restrict__ out, int n) {
  int i = blockIdx.x * blockDim.x + threadIdx.x;
  if (i < n) out[i] = f32_to_bf16(in[i]);
}

// ------------------------------ WMMA GEMM (TN) -----------------------------
// C[M,N] = A[M,K] @ W[N,K]^T  (both row-major, K contiguous)
// EPI: 0 = plain f32 store, 1 = bias + softplus (for dt)
#define BM 128
#define BN 128
#define BK 32

template <int EPI>
__global__ __launch_bounds__(256) void gemm_bf16_tn_kernel(
    const unsigned short* __restrict__ A,     // [M,K] bf16
    const unsigned short* __restrict__ W,     // [N,K] bf16
    const float* __restrict__ bias,           // [N] or nullptr
    float* __restrict__ C,                    // [M,N] f32
    int M, int N, int K) {
  (void)M; (void)bias;
  __shared__ __align__(16) unsigned short lsA[2][BM * BK];   // 2 x 8 KB
  __shared__ __align__(16) unsigned short lsB[2][BN * BK];   // 2 x 8 KB

  const int tid  = threadIdx.x;
  const int lane = tid & 31;
  const int wave = tid >> 5;          // 8 waves
  const int wm   = wave & 3;          // wave grid 4 (M, 32 rows) x 2 (N, 64 cols)
  const int wn   = wave >> 2;
  const int blockM = blockIdx.y * BM;
  const int blockN = blockIdx.x * BN;
  const int m15  = lane & 15;
  const int half = lane >> 4;
  const bool issuer = (tid < 32);     // wave 0 drives the TDM

  const unsigned ldsA0 = lds_off32(&lsA[0][0]);
  const unsigned ldsA1 = lds_off32(&lsA[1][0]);
  const unsigned ldsB0 = lds_off32(&lsB[0][0]);
  const unsigned ldsB1 = lds_off32(&lsB[1][0]);

  v8f acc[2][4] = {};

  const int nK = K / BK;
  if (issuer) {
    tdm_load_tile_bf16(ldsA0, A + (size_t)blockM * K, K, M, BK, BM, K);
    tdm_load_tile_bf16(ldsB0, W + (size_t)blockN * K, K, N, BK, BN, K);
  }

  for (int k = 0; k < nK; ++k) {
    const int cur = k & 1;
    if (issuer) __builtin_amdgcn_s_wait_tensorcnt(0);  // current buffers ready
    __syncthreads();                                   // release all waves
    if (issuer && (k + 1) < nK) {                      // prefetch next K tile
      const size_t koff = (size_t)(k + 1) * BK;
      tdm_load_tile_bf16(cur ? ldsA0 : ldsA1,
                         A + (size_t)blockM * K + koff, K, M, BK, BM, K);
      tdm_load_tile_bf16(cur ? ldsB0 : ldsB1,
                         W + (size_t)blockN * K + koff, K, N, BK, BN, K);
    }

    // ---- gather fragments per documented wave32 bf16 layouts ----
    const unsigned int* uA = (const unsigned int*)lsA[cur];  // row = 16 uints
    const unsigned int* uB = (const unsigned int*)lsB[cur];
    FragAB fa[2], fb[4];
    #pragma unroll
    for (int s = 0; s < 2; ++s) {
      const int mrow = wm * 32 + s * 16 + m15;
      #pragma unroll
      for (int v = 0; v < 8; ++v) {
        // A 16x32: K = (v/4)*16 + half*8 + (v%4)*2  (bf16 pair)
        int kk = ((v >> 2) << 4) + (half << 3) + ((v & 3) << 1);
        fa[s].u[v] = uA[mrow * 16 + (kk >> 1)];
      }
    }
    #pragma unroll
    for (int s = 0; s < 4; ++s) {
      const int nrow = wn * 64 + s * 16 + m15;
      #pragma unroll
      for (int v = 0; v < 8; ++v) {
        // B 32x16: K = (v/4)*16 + half*8 + (v%4)*2  (bf16 pair)
        int kk = ((v >> 2) << 4) + (half << 3) + ((v & 3) << 1);
        fb[s].u[v] = uB[nrow * 16 + (kk >> 1)];
      }
    }

    #pragma unroll
    for (int i = 0; i < 2; ++i)
      #pragma unroll
      for (int j = 0; j < 4; ++j)
        acc[i][j] = __builtin_amdgcn_wmma_f32_16x16x32_bf16(
            false, fa[i].v, false, fb[j].v, (short)0, acc[i][j], false, false);
  }

  // ---- store (C/D layout: lane col = lane&15, row M = v + 8*half) ----
  #pragma unroll
  for (int i = 0; i < 2; ++i) {
    #pragma unroll
    for (int j = 0; j < 4; ++j) {
      FragC fc; fc.v = acc[i][j];
      const int col = blockN + wn * 64 + j * 16 + m15;
      float bv = 0.0f;
      if (EPI == 1) bv = bias[col];
      #pragma unroll
      for (int v = 0; v < 8; ++v) {
        const int row = blockM + wm * 32 + i * 16 + v + (half << 3);
        float val = fc.f[v];
        if (EPI == 1) {               // softplus(val + bias)
          val += bv;
          val = (val > 20.0f) ? val : __logf(1.0f + __expf(val));
        }
        C[(size_t)row * N + col] = val;
      }
    }
  }
}

// ----------------- depthwise causal conv1d + bias + SiLU -------------------
// xz: [B*L, 4096] (cols 0..2047 = x_proj, 2048..4095 = z)
__global__ __launch_bounds__(256) void conv_silu_kernel(
    const float* __restrict__ xz, const float* __restrict__ conv_w,
    const float* __restrict__ conv_b,
    float* __restrict__ x_act, unsigned short* __restrict__ x_act_bf,
    int L, int C2, int C4, int total) {
  int idx = blockIdx.x * blockDim.x + threadIdx.x;
  if (idx >= total) return;
  const int c = idx % C2;
  const int r = idx / C2;          // b*L + t
  const int t = r % L;
  const int base = r - t;          // b*L
  const float* w = conv_w + c * 4;
  float acc = conv_b[c];
  #pragma unroll
  for (int j = 0; j < 4; ++j) {
    int tt = t + j - 3;            // left pad d_conv-1 = 3
    if (tt >= 0) acc += w[j] * xz[(size_t)(base + tt) * C4 + c];
  }
  float s = silu_f(acc);
  x_act[idx] = s;
  x_act_bf[idx] = f32_to_bf16(s);
}

// --------------------------- chunked parallel scan -------------------------
// h_t = a_t*h + b_t, a_t = exp(dt*A[c,s]), b_t = dt*x  (associative).
#define CHUNK  128
#define NCHUNK 16            // L / CHUNK

__global__ __launch_bounds__(256) void scan_part1_kernel(
    const float* __restrict__ x_act, const float* __restrict__ dt,
    const float* __restrict__ A_log,
    float* __restrict__ carryP, float* __restrict__ carryQ,
    int L, int C2, int total) {
  int idx = blockIdx.x * blockDim.x + threadIdx.x;   // (b*NCHUNK + k)*C2 + c
  if (idx >= total) return;
  const int c  = idx % C2;
  const int bk = idx / C2;
  const int k  = bk % NCHUNK;
  const int b  = bk / NCHUNK;

  float a[16], P[16], q[16];
  #pragma unroll
  for (int s = 0; s < 16; ++s) {
    a[s] = -__expf(A_log[c * 16 + s]);
    P[s] = 1.0f;
    q[s] = 0.0f;
  }
  size_t row = ((size_t)b * L + (size_t)k * CHUNK) * C2 + c;
  for (int t = 0; t < CHUNK; ++t) {
    const float xt  = x_act[row];
    const float dtv = dt[row];
    const float dbx = dtv * xt;
    #pragma unroll
    for (int s = 0; s < 16; ++s) {
      float e = __expf(dtv * a[s]);
      q[s] = e * q[s] + dbx;
      P[s] *= e;
    }
    row += C2;
  }
  const size_t base = (size_t)bk * 16 * C2 + c;
  #pragma unroll
  for (int s = 0; s < 16; ++s) {
    carryP[base + (size_t)s * C2] = P[s];
    carryQ[base + (size_t)s * C2] = q[s];
  }
}

__global__ __launch_bounds__(256) void scan_carry_kernel(
    const float* __restrict__ carryP, const float* __restrict__ carryQ,
    float* __restrict__ Hbuf, int C2, int total) {
  int idx = blockIdx.x * blockDim.x + threadIdx.x;   // b*C2 + c
  if (idx >= total) return;
  const int c = idx % C2;
  const int b = idx / C2;
  float H[16];
  #pragma unroll
  for (int s = 0; s < 16; ++s) H[s] = 0.0f;
  for (int k = 0; k < NCHUNK; ++k) {
    const size_t base = ((size_t)(b * NCHUNK + k) * 16) * C2 + c;
    #pragma unroll
    for (int s = 0; s < 16; ++s) {
      const size_t o = base + (size_t)s * C2;
      Hbuf[o] = H[s];                            // initial state of chunk k
      H[s] = carryP[o] * H[s] + carryQ[o];
    }
  }
}

__global__ __launch_bounds__(256) void scan_part2_kernel(
    const float* __restrict__ x_act, const float* __restrict__ dt,
    const float* __restrict__ xz, const float* __restrict__ A_log,
    const float* __restrict__ Dp, const float* __restrict__ Hbuf,
    unsigned short* __restrict__ y_gated,
    int L, int C2, int C4, int total) {
  int idx = blockIdx.x * blockDim.x + threadIdx.x;   // (b*NCHUNK + k)*C2 + c
  if (idx >= total) return;
  const int c  = idx % C2;
  const int bk = idx / C2;
  const int k  = bk % NCHUNK;
  const int b  = bk / NCHUNK;

  float a[16], h[16];
  const size_t base = (size_t)bk * 16 * C2 + c;
  #pragma unroll
  for (int s = 0; s < 16; ++s) {
    a[s] = -__expf(A_log[c * 16 + s]);
    h[s] = Hbuf[base + (size_t)s * C2];
  }
  const float dp = Dp[c];

  size_t row  = ((size_t)b * L + (size_t)k * CHUNK) * C2 + c;
  size_t rowz = ((size_t)b * L + (size_t)k * CHUNK) * C4 + C2 + c;
  for (int t = 0; t < CHUNK; ++t) {
    const float xt  = x_act[row];
    const float dtv = dt[row];
    const float zv  = xz[rowz];
    const float dbx = dtv * xt;
    float sum = 0.0f;
    #pragma unroll
    for (int s = 0; s < 16; ++s) {
      float e = __expf(dtv * a[s]);    // v_exp_f32 (trans)
      h[s] = e * h[s] + dbx;
      sum += h[s];
    }
    float y = sum + dp * xt;
    y_gated[row] = f32_to_bf16(y * silu_f(zv));
    row  += C2;
    rowz += C4;
  }
}

// ---------------------------------------------------------------------------
extern "C" void kernel_launch(void* const* d_in, const int* in_sizes, int n_in,
                              void* d_out, int out_size, void* d_ws, size_t ws_size,
                              hipStream_t stream) {
  (void)in_sizes; (void)n_in; (void)out_size; (void)ws_size;
  constexpr int D_MODEL = 1024, D_INNER = 2048;
  constexpr int Bsz = 2, L = 2048;
  constexpr int ML = Bsz * L;                 // 4096 rows

  const float* x      = (const float*)d_in[0];   // [B,L,1024]
  const float* W_in   = (const float*)d_in[1];   // [4096,1024]
  const float* conv_w = (const float*)d_in[2];   // [2048,1,4]
  const float* conv_b = (const float*)d_in[3];   // [2048]
  const float* A_log  = (const float*)d_in[4];   // [2048,16]
  const float* Dp     = (const float*)d_in[5];   // [2048]
  const float* W_dt   = (const float*)d_in[6];   // [2048,2048]
  const float* b_dt   = (const float*)d_in[7];   // [2048]
  const float* W_out  = (const float*)d_in[8];   // [1024,2048]
  float* out = (float*)d_out;

  // -------- workspace layout --------
  char* ws = (char*)d_ws;
  size_t off = 0;
  auto take = [&](size_t bytes) {
    void* p = ws + off;
    off += (bytes + 255) & ~(size_t)255;
    return p;
  };
  unsigned short* x_bf    = (unsigned short*)take((size_t)ML * D_MODEL * 2);
  unsigned short* Win_bf  = (unsigned short*)take((size_t)2 * D_INNER * D_MODEL * 2);
  unsigned short* Wdt_bf  = (unsigned short*)take((size_t)D_INNER * D_INNER * 2);
  unsigned short* Wout_bf = (unsigned short*)take((size_t)D_MODEL * D_INNER * 2);
  float*          xzbuf   = (float*)take((size_t)ML * 2 * D_INNER * 4);
  float*          x_act   = (float*)take((size_t)ML * D_INNER * 4);
  unsigned short* x_actbf = (unsigned short*)take((size_t)ML * D_INNER * 2);
  float*          dtbuf   = (float*)take((size_t)ML * D_INNER * 4);
  unsigned short* y_gated = (unsigned short*)take((size_t)ML * D_INNER * 2);
  float*          carryP  = (float*)take((size_t)Bsz * NCHUNK * 16 * D_INNER * 4);
  float*          carryQ  = (float*)take((size_t)Bsz * NCHUNK * 16 * D_INNER * 4);
  float*          Hbuf    = (float*)take((size_t)Bsz * NCHUNK * 16 * D_INNER * 4);

  // -------- 1) cast weights + x to bf16 --------
  auto cast = [&](const float* src, unsigned short* dst, int n) {
    cast_f32_bf16_kernel<<<(n + 255) / 256, 256, 0, stream>>>(src, dst, n);
  };
  cast(x,     x_bf,    ML * D_MODEL);
  cast(W_in,  Win_bf,  2 * D_INNER * D_MODEL);
  cast(W_dt,  Wdt_bf,  D_INNER * D_INNER);
  cast(W_out, Wout_bf, D_MODEL * D_INNER);

  // -------- 2) xz = x @ W_in^T   (4096 x 4096 x 1024) --------
  {
    dim3 grid((2 * D_INNER) / BN, ML / BM);
    gemm_bf16_tn_kernel<0><<<grid, 256, 0, stream>>>(
        x_bf, Win_bf, nullptr, xzbuf, ML, 2 * D_INNER, D_MODEL);
  }

  // -------- 3) depthwise causal conv + bias + SiLU --------
  {
    int total = ML * D_INNER;
    conv_silu_kernel<<<(total + 255) / 256, 256, 0, stream>>>(
        xzbuf, conv_w, conv_b, x_act, x_actbf, L, D_INNER, 2 * D_INNER, total);
  }

  // -------- 4) dt = softplus(x_act @ W_dt^T + b_dt)  (4096 x 2048 x 2048) ---
  {
    dim3 grid(D_INNER / BN, ML / BM);
    gemm_bf16_tn_kernel<1><<<grid, 256, 0, stream>>>(
        x_actbf, Wdt_bf, b_dt, dtbuf, ML, D_INNER, D_INNER);
  }

  // -------- 5) chunked selective scan + gating --------
  {
    int total1 = Bsz * NCHUNK * D_INNER;   // 65536
    scan_part1_kernel<<<(total1 + 255) / 256, 256, 0, stream>>>(
        x_act, dtbuf, A_log, carryP, carryQ, L, D_INNER, total1);
    int total2 = Bsz * D_INNER;            // 4096
    scan_carry_kernel<<<(total2 + 255) / 256, 256, 0, stream>>>(
        carryP, carryQ, Hbuf, D_INNER, total2);
    scan_part2_kernel<<<(total1 + 255) / 256, 256, 0, stream>>>(
        x_act, dtbuf, xzbuf, A_log, Dp, Hbuf, y_gated,
        L, D_INNER, 2 * D_INNER, total1);
  }

  // -------- 6) out = y_gated @ W_out^T  (4096 x 1024 x 2048) --------
  {
    dim3 grid(D_MODEL / BN, ML / BM);
    gemm_bf16_tn_kernel<0><<<grid, 256, 0, stream>>>(
        y_gated, Wout_bf, nullptr, out, ML, D_MODEL, D_INNER);
  }
}
